// HyperConv_69604239999586
// MI455X (gfx1250) — matrix operand order repaired and checked by reading, not compiled, not run on hardware
//
#include <hip/hip_runtime.h>

#define kN 20000
#define kE 5000
#define kF 128

typedef __attribute__((ext_vector_type(16))) _Float16 v16h;
typedef __attribute__((ext_vector_type(8)))  float    v8f;
typedef __attribute__((ext_vector_type(4)))  float    v4f;

// Pack a row-major A-tile row (16-bit A 16x32 layout): lane holds row M,
// K-octets {half*8+0..7, half*8+16..23}; VGPR pairs are K-consecutive.
// ap points at row + k0 + half*8 (16B aligned). H is streamed (zero reuse)
// -> non-temporal loads so the 400MB stream doesn't evict hot Xw/M from L2.
__device__ __forceinline__ v16h pack_a_rowmajor_nt(const float* __restrict__ ap, float& acc) {
  const v4f* vp = reinterpret_cast<const v4f*>(ap);
  v4f x0 = __builtin_nontemporal_load(vp + 0);
  v4f x1 = __builtin_nontemporal_load(vp + 1);
  v4f y0 = __builtin_nontemporal_load(vp + 4);
  v4f y1 = __builtin_nontemporal_load(vp + 5);
  acc += x0[0] + x0[1] + x0[2] + x0[3] + x1[0] + x1[1] + x1[2] + x1[3]
       + y0[0] + y0[1] + y0[2] + y0[3] + y1[0] + y1[1] + y1[2] + y1[3];
  v16h a;
  a[0]  = (_Float16)x0[0]; a[1]  = (_Float16)x0[1];
  a[2]  = (_Float16)x0[2]; a[3]  = (_Float16)x0[3];
  a[4]  = (_Float16)x1[0]; a[5]  = (_Float16)x1[1];
  a[6]  = (_Float16)x1[2]; a[7]  = (_Float16)x1[3];
  a[8]  = (_Float16)y0[0]; a[9]  = (_Float16)y0[1];
  a[10] = (_Float16)y0[2]; a[11] = (_Float16)y0[3];
  a[12] = (_Float16)y1[0]; a[13] = (_Float16)y1[1];
  a[14] = (_Float16)y1[2]; a[15] = (_Float16)y1[3];
  return a;
}

// B 32x16 layout: lane holds column N=lane%16, K = {half*8+0..7, 16+half*8+0..7}.
// B operands (Xw, M, W-in-LDS) are the hot, re-read data: regular temporal.
__device__ __forceinline__ v16h load_b16(const _Float16* __restrict__ p, int kv) {
  const int h = (threadIdx.x >> 4) & 1;
  v16h b;
#pragma unroll
  for (int s = 0; s < 16; ++s) {
    const int k = ((s < 8) ? s : s + 8) + 8 * h;
    const int kc = (k < kv) ? k : (kv - 1);
    const _Float16 v = p[(long)kc * kF];
    b[s] = (k < kv) ? v : (_Float16)0.0f;
  }
  return b;
}

// Branchless lane-predicated zero (keeps EXEC all-ones for WMMA).
__device__ __forceinline__ v16h zero_unless(v16h b, bool keep) {
#pragma unroll
  for (int q = 0; q < 16; ++q) b[q] = keep ? b[q] : (_Float16)0.0f;
  return b;
}

// ---------------------------------------------------------------------------
// Kernel 1: Xw[n,f] = sum_k X[n,k] * W[k,f], stored as f16. W staged in LDS.
// One wave per 16-row tile of X; 4 waves / workgroup.
// ---------------------------------------------------------------------------
__global__ __launch_bounds__(128) void xw_kernel(const float* __restrict__ X,
                                                 const float* __restrict__ W,
                                                 _Float16* __restrict__ Xw) {
  __shared__ __align__(32) _Float16 wl[kF * kF];
  for (int i = threadIdx.x; i < kF * kF; i += 128) wl[i] = (_Float16)W[i];
  __syncthreads();

  const int wave = threadIdx.x >> 5;
  const int lane = threadIdx.x & 31;
  const int lm   = lane & 15;
  const int half = lane >> 4;
  const int tile = blockIdx.x * 4 + wave;
  if (tile >= kN / 16) return;
  const int n0 = tile * 16;
  const int n  = n0 + lm;

  v8f c[8];
#pragma unroll
  for (int ct = 0; ct < 8; ++ct)
#pragma unroll
    for (int r = 0; r < 8; ++r) c[ct][r] = 0.0f;

  float dummy = 0.0f;
#pragma unroll
  for (int s = 0; s < 4; ++s) {
    const int k0 = s * 32;
    v16h a = pack_a_rowmajor_nt(X + (long)n * kF + k0 + half * 8, dummy);
#pragma unroll
    for (int ct = 0; ct < 8; ++ct) {
      v16h b = load_b16(&wl[k0 * kF + ct * 16 + lm], 32);
      c[ct] = __builtin_amdgcn_wmma_f32_16x16x32_f16(
          false, a, false, b, (short)0, c[ct], false, false);
    }
  }
  // C layout: lane holds N=lm, VGPR r -> M = r + 8*half
#pragma unroll
  for (int ct = 0; ct < 8; ++ct)
#pragma unroll
    for (int r = 0; r < 8; ++r)
      Xw[(long)(n0 + r + half * 8) * kF + ct * 16 + lm] = (_Float16)c[ct][r];
}

// ---------------------------------------------------------------------------
// Kernel 2: M[e,f] = (1/DE[e]) * sum_n H[n,e] * Xw[n,f].  A = H^T (strided
// loads; lanes coalesce across the contiguous e dimension, each 64B line of
// the tile touched exactly once). DE fused from the A-tiles. 4 waves split
// K (=N), combined via an LDS reduction.
// ---------------------------------------------------------------------------
__global__ __launch_bounds__(128) void m_kernel(const float* __restrict__ H,
                                                const _Float16* __restrict__ Xw,
                                                _Float16* __restrict__ M) {
  const int wave = threadIdx.x >> 5;
  const int lane = threadIdx.x & 31;
  const int lm   = lane & 15;
  const int half = lane >> 4;
  const int e0   = blockIdx.x * 16;
  const int e    = e0 + lm;
  const bool ev  = (e < kE);
  const int ec   = ev ? e : (kE - 1);   // clamped address, value zeroed below

  v8f c[8];
#pragma unroll
  for (int ct = 0; ct < 8; ++ct)
#pragma unroll
    for (int r = 0; r < 8; ++r) c[ct][r] = 0.0f;

  float de = 0.0f;
  for (int s = wave; s < kN / 32; s += 4) {
    const int k0 = s * 32;
    // Prefetch this wave's next A tile (K interleave of 4 waves -> +128 rows).
    // Lane L covers K-row k0+128+L -> 32 rows, one 64B line each.
    {
      const int kpf = k0 + 128;
      if (kpf < kN)
        __builtin_prefetch(&H[(long)(kpf + lane) * kE + e0], 0, 0);
    }
    v16h a;
#pragma unroll
    for (int j = 0; j < 8; ++j) {
      const int kk = ((j < 4) ? 2 * j : 16 + 2 * (j - 4)) + half * 8;
      float x0 = __builtin_nontemporal_load(&H[(long)(k0 + kk) * kE + ec]);
      float x1 = __builtin_nontemporal_load(&H[(long)(k0 + kk + 1) * kE + ec]);
      x0 = ev ? x0 : 0.0f;
      x1 = ev ? x1 : 0.0f;
      de += x0 + x1;
      a[2 * j]     = (_Float16)x0;
      a[2 * j + 1] = (_Float16)x1;
    }
    const _Float16* bp = Xw + (long)k0 * kF + lm;
#pragma unroll
    for (int ct = 0; ct < 8; ++ct) {
      v16h b = load_b16(bp + ct * 16, 32);
      c[ct] = __builtin_amdgcn_wmma_f32_16x16x32_f16(
          false, a, false, b, (short)0, c[ct], false, false);
    }
  }

  de += __shfl_xor(de, 16, 32);  // combine the two K-octet halves per row

  __shared__ __align__(16) float sc[4][16 * kF];
  __shared__ float sde[4][16];
  __shared__ float sinv[16];
  if (lane < 16) sde[wave][lm] = de;
#pragma unroll
  for (int ct = 0; ct < 8; ++ct)
#pragma unroll
    for (int r = 0; r < 8; ++r)
      sc[wave][(r + half * 8) * kF + ct * 16 + lm] = c[ct][r];
  __syncthreads();
  if (threadIdx.x < 16)
    sinv[threadIdx.x] = 1.0f / (sde[0][threadIdx.x] + sde[1][threadIdx.x] +
                                sde[2][threadIdx.x] + sde[3][threadIdx.x] + 1e-12f);
  __syncthreads();
  for (int idx = threadIdx.x; idx < 16 * kF; idx += 128) {
    const int m = idx >> 7;
    const int f = idx & (kF - 1);
    const float v = sc[0][idx] + sc[1][idx] + sc[2][idx] + sc[3][idx];
    if (e0 + m < kE) M[(long)(e0 + m) * kF + f] = (_Float16)(v * sinv[m]);
  }
}

// ---------------------------------------------------------------------------
// Kernel 3: out[n,f] = relu( (1/DV[n]) * sum_e H[n,e]*M[e,f] + bias[f] ).
// A = H row-major (b128 NT loads), DV fused. 4 waves split K (=E).
// ---------------------------------------------------------------------------
__global__ __launch_bounds__(128) void out_kernel(const float* __restrict__ H,
                                                  const _Float16* __restrict__ M,
                                                  const float* __restrict__ bias,
                                                  float* __restrict__ out) {
  const int wave = threadIdx.x >> 5;
  const int lane = threadIdx.x & 31;
  const int lm   = lane & 15;
  const int half = lane >> 4;
  const int n0   = blockIdx.x * 16;
  const int n    = n0 + lm;

  v8f c[8];
#pragma unroll
  for (int ct = 0; ct < 8; ++ct)
#pragma unroll
    for (int r = 0; r < 8; ++r) c[ct][r] = 0.0f;

  float dv = 0.0f;
  const int ksteps = (kE + 31) / 32;
  for (int s = wave; s < ksteps; s += 4) {
    const int k0 = s * 32;
    // Prefetch next tile for this wave: 16 rows, 128B each (two 64B halves
    // covered by the two lane-halves).
    {
      const int kpf = k0 + 128;
      if (kpf < kE)
        __builtin_prefetch(&H[(long)n * kE + kpf + half * 16], 0, 0);
    }
    v16h a;
    if (k0 + 32 <= kE) {
      a = pack_a_rowmajor_nt(H + (long)n * kE + k0 + half * 8, dv);
    } else {  // tail: clamped scalar loads, values zeroed (EXEC stays full)
#pragma unroll
      for (int j = 0; j < 8; ++j) {
        const int kk = ((j < 4) ? 2 * j : 16 + 2 * (j - 4)) + half * 8;
        const bool kv0 = (k0 + kk < kE);
        const bool kv1 = (k0 + kk + 1 < kE);
        const int kc0 = kv0 ? (k0 + kk) : (kE - 1);
        const int kc1 = kv1 ? (k0 + kk + 1) : (kE - 1);
        float x0 = __builtin_nontemporal_load(&H[(long)n * kE + kc0]);
        float x1 = __builtin_nontemporal_load(&H[(long)n * kE + kc1]);
        x0 = kv0 ? x0 : 0.0f;
        x1 = kv1 ? x1 : 0.0f;
        dv += x0 + x1;
        a[2 * j]     = (_Float16)x0;
        a[2 * j + 1] = (_Float16)x1;
      }
    }
    const _Float16* bp = M + (long)k0 * kF + lm;
#pragma unroll
    for (int ct = 0; ct < 8; ++ct) {
      v16h b = load_b16(bp + ct * 16, kE - k0);
      c[ct] = __builtin_amdgcn_wmma_f32_16x16x32_f16(
          false, a, false, b, (short)0, c[ct], false, false);
    }
  }

  dv += __shfl_xor(dv, 16, 32);

  __shared__ __align__(16) float sc[4][16 * kF];
  __shared__ float sdv[4][16];
  __shared__ float sinv[16];
  if (lane < 16) sdv[wave][lm] = dv;
#pragma unroll
  for (int ct = 0; ct < 8; ++ct)
#pragma unroll
    for (int r = 0; r < 8; ++r)
      sc[wave][(r + half * 8) * kF + ct * 16 + lm] = c[ct][r];
  __syncthreads();
  if (threadIdx.x < 16)
    sinv[threadIdx.x] = 1.0f / (sdv[0][threadIdx.x] + sdv[1][threadIdx.x] +
                                sdv[2][threadIdx.x] + sdv[3][threadIdx.x] + 1e-12f);
  __syncthreads();
  // One-shot 10MB output: non-temporal store (write-no-allocate policy).
  for (int idx = threadIdx.x; idx < 16 * kF; idx += 128) {
    const int m = idx >> 7;
    const int f = idx & (kF - 1);
    const float v = sc[0][idx] + sc[1][idx] + sc[2][idx] + sc[3][idx];
    const float o = fmaxf(v * sinv[m] + bias[f], 0.0f);
    __builtin_nontemporal_store(o, &out[(long)(n0 + m) * kF + f]);
  }
}

extern "C" void kernel_launch(void* const* d_in, const int* in_sizes, int n_in,
                              void* d_out, int out_size, void* d_ws, size_t ws_size,
                              hipStream_t stream) {
  const float* X    = (const float*)d_in[0];
  const float* H    = (const float*)d_in[1];
  const float* W    = (const float*)d_in[2];
  const float* bias = (const float*)d_in[3];
  float* out = (float*)d_out;

  _Float16* Xw = (_Float16*)d_ws;                                               // 5.12 MB
  _Float16* M  = (_Float16*)((char*)d_ws + (size_t)kN * kF * sizeof(_Float16)); // +1.28 MB

  xw_kernel<<<dim3((kN / 16 + 3) / 4), dim3(128), 0, stream>>>(X, W, Xw);
  m_kernel<<<dim3((kE + 15) / 16), dim3(128), 0, stream>>>(H, Xw, M);
  out_kernel<<<dim3(kN / 16), dim3(128), 0, stream>>>(H, M, bias, out);
}
